// SingleStreamBlock_53790170415362
// MI455X (gfx1250) — matrix operand order repaired and checked by reading, not compiled, not run on hardware
//
#include <hip/hip_runtime.h>
#include <hip/hip_bf16.h>

// ---------------------------------------------------------------------------
// SingleStreamBlock for MI455X (gfx1250, wave32, WMMA bf16, async->LDS)
// ---------------------------------------------------------------------------

#define L_SEQ   3072
#define HID     3072
#define NHEAD   24
#define DHEAD   128
#define MLP_D   12288
#define N1      (3*HID + MLP_D)     // 21504 : lin1 output cols
#define K2      (HID + MLP_D)       // 15360 : lin2 inner dim
#define EPSF    1e-6f

typedef __attribute__((ext_vector_type(16))) __bf16 v16bf;
typedef __attribute__((ext_vector_type(8)))  float  v8f;

union Frag { uint4 u4[2]; v16bf v; };

__device__ __forceinline__ v8f wmma_bf16(v16bf a, v16bf b, v8f c) {
  return __builtin_amdgcn_wmma_f32_16x16x32_bf16(false, a, false, b, (short)0, c,
                                                 false, false);
}

// CDNA5 async global->LDS copy, 16B per lane, tracked by ASYNCcnt.
__device__ __forceinline__ void async_copy16(unsigned lds_addr, const void* g) {
  asm volatile("global_load_async_to_lds_b128 %0, %1, off"
               :: "v"(lds_addr), "v"(g)
               : "memory");
}
__device__ __forceinline__ void async_wait0() {
  asm volatile("s_wait_asynccnt 0x0" ::: "memory");
}
__device__ __forceinline__ unsigned lds_addr_of(const void* p) {
  // flat LDS addresses carry the LDS byte offset in the low 32 bits (ISA 10.2)
  return (unsigned)(uintptr_t)p;
}

// A-matrix fragment (16x32 bf16): lane m = lane&15 holds 16 contiguous K
// [half*16, half*16+16): two 16B loads.
__device__ __forceinline__ v16bf fragA(const __bf16* rowBase, int half) {
  Frag f;
  f.u4[0] = *(const uint4*)(rowBase + half * 16);
  f.u4[1] = *(const uint4*)(rowBase + half * 16 + 8);
  return f.v;
}
// B-matrix fragment (32x16 bf16): lane n = lane&15 holds 16 contiguous K at
// offset half*16: one 32B (two 16B) load. Caller passes lane's 32B start.
__device__ __forceinline__ v16bf fragB(const __bf16* p) {
  Frag f;
  f.u4[0] = *(const uint4*)(p);
  f.u4[1] = *(const uint4*)(p + 8);
  return f.v;
}

__device__ __forceinline__ float gelu_tanh(float x) {
  float c = 0.7978845608028654f * (x + 0.044715f * x * x * x);
  return 0.5f * x * (1.0f + tanhf(c));
}

// f32 -> bf16 convert + store 16 values to LDS (weight staging path)
__device__ __forceinline__ void storeB(__bf16* Bs, int lrow, int lhalf,
                                       const float4* fb) {
  union { __bf16 h[16]; uint4 u[2]; } pk;
  const float* ff = (const float*)fb;
#pragma unroll
  for (int i = 0; i < 16; i++) pk.h[i] = (__bf16)ff[i];
  __bf16* pb = Bs + lrow * 40 + lhalf * 16;
  *(uint4*)pb = pk.u[0]; *(uint4*)(pb + 8) = pk.u[1];
}

// ---------------------------------------------------------------------------
// Kernel 1: mod = silu(vec) @ mod_w.T + mod_b     (9216 outputs)
// ---------------------------------------------------------------------------
__global__ __launch_bounds__(256) void mod_gemv_kernel(
    const float* __restrict__ vec, const float* __restrict__ mw,
    const float* __restrict__ mb, float* __restrict__ modo) {
  __shared__ float sv[HID];
  int tid = threadIdx.x;
  for (int i = tid; i < HID; i += 256) {
    float v = vec[i];
    sv[i] = v / (1.0f + __expf(-v));
  }
  __syncthreads();
  int lane = tid & 31;
  int row = blockIdx.x * 8 + (tid >> 5);
  const float* wr = mw + (size_t)row * HID;
  float s = 0.f;
  for (int c = lane; c < HID; c += 32) s += sv[c] * wr[c];
  for (int off = 16; off; off >>= 1) s += __shfl_xor(s, off);
  if (lane == 0) modo[row] = s + mb[row];
}

// ---------------------------------------------------------------------------
// Kernel 2: modulated LayerNorm, write bf16 x_mod
// ---------------------------------------------------------------------------
__global__ __launch_bounds__(256) void ln_mod_kernel(
    const float* __restrict__ x, const float* __restrict__ mod,
    __bf16* __restrict__ xmod) {
  int l = blockIdx.x, tid = threadIdx.x;
  int lane = tid & 31, w = tid >> 5;
  const float* xr = x + (size_t)l * HID;
  float xa[12], s = 0.f, s2 = 0.f;
#pragma unroll
  for (int i = 0; i < 12; i++) {
    float v = xr[tid + i * 256];
    xa[i] = v; s += v; s2 += v * v;
  }
  for (int off = 16; off; off >>= 1) { s += __shfl_xor(s, off); s2 += __shfl_xor(s2, off); }
  __shared__ float red[2][8];
  if (lane == 0) { red[0][w] = s; red[1][w] = s2; }
  __syncthreads();
  float ts = 0.f, ts2 = 0.f;
#pragma unroll
  for (int j = 0; j < 8; j++) { ts += red[0][j]; ts2 += red[1][j]; }
  float mu = ts * (1.0f / HID);
  float var = ts2 * (1.0f / HID) - mu * mu;
  float rr = rsqrtf(var + EPSF);
  __bf16* orow = xmod + (size_t)l * HID;
#pragma unroll
  for (int i = 0; i < 12; i++) {
    int c = tid + i * 256;
    float sc = 1.0f + mod[HID + c];
    float sh = mod[c];
    orow[c] = (__bf16)((xa[i] - mu) * rr * sc + sh);
  }
}

// ---------------------------------------------------------------------------
// Shared GEMM machinery: C(128x128 per block) = A(bf16, MxK) * B(f32, NxK)^T
// A-tiles: GLOBAL_LOAD_ASYNC_TO_LDS (no VGPR round trip, ASYNCcnt).
// B-tiles: f32 global -> v_cvt_pk_bf16 -> LDS. Double buffered.
// ---------------------------------------------------------------------------
#define GEMM_MAINLOOP(KDIM)                                                    \
  const int KT = (KDIM) / 32;                                                  \
  __shared__ __bf16 lds[2][2][128 * 40];                                       \
  int tid = threadIdx.x, lane = tid & 31, wv = tid >> 5;                       \
  int wm = wv & 1, wn = wv >> 1;                                               \
  int half = lane >> 4, col = lane & 15;                                       \
  int mBase = blockIdx.x * 128, nBase = blockIdx.y * 128;                      \
  int lrow = tid >> 1, lhalf = tid & 1;                                        \
  const __bf16* gA = A + (size_t)(mBase + lrow) * (KDIM) + lhalf * 16;         \
  const float*  gW = W + (size_t)(nBase + lrow) * (KDIM) + lhalf * 16;         \
  unsigned ldsA[2];                                                            \
  ldsA[0] = lds_addr_of(&lds[0][0][0] + lrow * 40 + lhalf * 16);               \
  ldsA[1] = lds_addr_of(&lds[1][0][0] + lrow * 40 + lhalf * 16);               \
  v8f acc[4][2];                                                               \
  _Pragma("unroll") for (int i = 0; i < 4; i++)                                \
  _Pragma("unroll") for (int j = 0; j < 2; j++)                                \
      acc[i][j] = (v8f){0.f,0.f,0.f,0.f,0.f,0.f,0.f,0.f};                      \
  {                                                                            \
    async_copy16(ldsA[0], gA);                                                 \
    async_copy16(ldsA[0] + 16, gA + 8);                                        \
    const float4* gB = (const float4*)gW;                                      \
    float4 fb[4] = {gB[0], gB[1], gB[2], gB[3]};                               \
    storeB(&lds[0][1][0], lrow, lhalf, fb);                                    \
  }                                                                            \
  for (int kt = 0; kt < KT; ++kt) {                                            \
    int buf = kt & 1;                                                          \
    async_wait0();                                                             \
    __syncthreads();                                                           \
    bool more = (kt + 1 < KT);                                                 \
    float4 nb[4];                                                              \
    if (more) {                                                                \
      const __bf16* pA = gA + (kt + 1) * 32;                                   \
      async_copy16(ldsA[buf ^ 1], pA);                                         \
      async_copy16(ldsA[buf ^ 1] + 16, pA + 8);                                \
      const float4* gB = (const float4*)(gW + (kt + 1) * 32);                  \
      nb[0] = gB[0]; nb[1] = gB[1]; nb[2] = gB[2]; nb[3] = gB[3];              \
    }                                                                          \
    if (kt + 2 < KT) __builtin_prefetch(gW + (kt + 2) * 32, 0, 0);             \
    const __bf16* As = &lds[buf][0][0];                                        \
    const __bf16* Bs = &lds[buf][1][0];                                        \
    v16bf av[4], bvv[2];                                                       \
    _Pragma("unroll") for (int i = 0; i < 4; i++)                              \
        av[i] = fragA(As + (wm * 64 + i * 16 + col) * 40, half);               \
    _Pragma("unroll") for (int j = 0; j < 2; j++)                              \
        bvv[j] = fragB(Bs + (wn * 32 + j * 16 + col) * 40 + half * 16);        \
    _Pragma("unroll") for (int i = 0; i < 4; i++)                              \
    _Pragma("unroll") for (int j = 0; j < 2; j++)                              \
        acc[i][j] = wmma_bf16(av[i], bvv[j], acc[i][j]);                       \
    if (more) storeB(&lds[buf ^ 1][1][0], lrow, lhalf, nb);                    \
  }

// ---------------------------------------------------------------------------
// Kernel 3: GEMM1: proj = x_mod @ lin1_w.T + b ; scatter q/k/v + gelu(mlp)
//   q,k -> [h][l][d] bf16 ;  v -> [h][d][l] bf16 ;  mlp -> cat[:, 3072:]
// ---------------------------------------------------------------------------
__global__ __launch_bounds__(256) void gemm1_kernel(
    const __bf16* __restrict__ A, const float* __restrict__ W,
    const float* __restrict__ bias, __bf16* __restrict__ qb,
    __bf16* __restrict__ kb, __bf16* __restrict__ vb,
    __bf16* __restrict__ cat) {
  GEMM_MAINLOOP(HID)
#pragma unroll
  for (int i = 0; i < 4; i++) {
#pragma unroll
    for (int j = 0; j < 2; j++) {
      int n = nBase + wn * 32 + j * 16 + col;
      float bi = bias[n];
#pragma unroll
      for (int r = 0; r < 8; r++) {
        int l = mBase + wm * 64 + i * 16 + r + half * 8;
        float val = acc[i][j][r] + bi;
        if (n < 3 * HID) {
          int which = n / HID, idx = n % HID;
          int hh = idx >> 7, d = idx & 127;
          if (which == 2)
            vb[((size_t)hh * DHEAD + d) * L_SEQ + l] = (__bf16)val;
          else {
            __bf16* dst = which ? kb : qb;
            dst[((size_t)hh * L_SEQ + l) * DHEAD + d] = (__bf16)val;
          }
        } else {
          cat[(size_t)l * K2 + HID + (n - 3 * HID)] = (__bf16)gelu_tanh(val);
        }
      }
    }
  }
}

// ---------------------------------------------------------------------------
// Kernel 4: per (l,h,{q|k}): RMS-norm(scale) + RoPE, in place. q gets 1/sqrt(D).
// ---------------------------------------------------------------------------
__global__ __launch_bounds__(128) void rope_rms_kernel(
    __bf16* __restrict__ q, __bf16* __restrict__ k,
    const float* __restrict__ pe, const float* __restrict__ qs,
    const float* __restrict__ ks) {
  int l = blockIdx.x, hh = blockIdx.y, isk = blockIdx.z;
  __bf16* buf = (isk ? k : q) + ((size_t)hh * L_SEQ + l) * DHEAD;
  const float* scl = isk ? ks : qs;
  float sf = isk ? 1.0f : 0.08838834764831843f;  // 1/sqrt(128) folded into q
  int d = threadIdx.x, lane = d & 31, w = d >> 5;
  float xv = (float)buf[d];
  float s2 = xv * xv;
  for (int off = 16; off; off >>= 1) s2 += __shfl_xor(s2, off);
  __shared__ float r4[4];
  __shared__ float tv[DHEAD];
  if (lane == 0) r4[w] = s2;
  __syncthreads();
  float tot = r4[0] + r4[1] + r4[2] + r4[3];
  float rr = rsqrtf(tot * (1.0f / DHEAD) + EPSF);
  tv[d] = xv * rr * scl[d];
  __syncthreads();
  if (d < 64) {
    const float* p4 = pe + (size_t)l * 256 + d * 4;
    float t0 = tv[2 * d], t1 = tv[2 * d + 1];
    buf[2 * d]     = (__bf16)(sf * (p4[0] * t0 + p4[1] * t1));
    buf[2 * d + 1] = (__bf16)(sf * (p4[2] * t0 + p4[3] * t1));
  }
}

// ---------------------------------------------------------------------------
// Kernel 5: flash attention. Block = (head h, 128-query tile). 8 waves,
// wave handles 16 query rows. Key tiles of 32, double-buffered via
// GLOBAL_LOAD_ASYNC_TO_LDS, online softmax, WMMA bf16.
// Output -> cat[:, h*128 + d]  (first HID columns of cat).
// ---------------------------------------------------------------------------
__global__ __launch_bounds__(256) void attn_kernel(
    const __bf16* __restrict__ q, const __bf16* __restrict__ k,
    const __bf16* __restrict__ vt, __bf16* __restrict__ cat) {
  int h = blockIdx.y, qt = blockIdx.x;
  int tid = threadIdx.x, lane = tid & 31, w = tid >> 5;
  int half = lane >> 4, col = lane & 15;
  int q0 = qt * 128 + w * 16;

  __shared__ __bf16 Ks[2][32 * 136];   // [key][d], pitch 136 bf16
  __shared__ __bf16 Vs[2][128 * 40];   // [d][key], pitch 40 bf16
  __shared__ __bf16 Ps[8 * 16 * 40];   // per-wave 16x32 P, pitch 40

  // resident Q fragments (16 rows x 128 d)
  v16bf qa[4];
  {
    const __bf16* qrow = q + ((size_t)h * L_SEQ + q0 + col) * DHEAD;
#pragma unroll
    for (int kc = 0; kc < 4; kc++) qa[kc] = fragA(qrow + kc * 32, half);
  }
  v8f o[8];
#pragma unroll
  for (int t = 0; t < 8; t++) o[t] = (v8f){0.f,0.f,0.f,0.f,0.f,0.f,0.f,0.f};
  float mi[8], li[8];
#pragma unroll
  for (int r = 0; r < 8; r++) { mi[r] = -1e30f; li[r] = 0.f; }

  int krow = tid >> 3, kseg = (tid & 7) * 16;
  int vd = tid >> 1, vh = (tid & 1) * 16;
  const __bf16* gK = k + ((size_t)h * L_SEQ + krow) * DHEAD + kseg;
  const __bf16* gV = vt + ((size_t)h * DHEAD + vd) * L_SEQ + vh;
  unsigned lK[2], lV[2];
  lK[0] = lds_addr_of(&Ks[0][krow * 136 + kseg]);
  lK[1] = lds_addr_of(&Ks[1][krow * 136 + kseg]);
  lV[0] = lds_addr_of(&Vs[0][vd * 40 + vh]);
  lV[1] = lds_addr_of(&Vs[1][vd * 40 + vh]);

  // prologue: async-stage tile 0
  async_copy16(lK[0], gK);      async_copy16(lK[0] + 16, gK + 8);
  async_copy16(lV[0], gV);      async_copy16(lV[0] + 16, gV + 8);

  const int NT = L_SEQ / 32;
  for (int kt = 0; kt < NT; ++kt) {
    int buf = kt & 1;
    async_wait0();
    __syncthreads();
    if (kt + 1 < NT) {  // async-stage next tile while we compute
      const __bf16* pK = gK + (size_t)(kt + 1) * 32 * DHEAD;
      const __bf16* pV = gV + (kt + 1) * 32;
      async_copy16(lK[buf ^ 1], pK);  async_copy16(lK[buf ^ 1] + 16, pK + 8);
      async_copy16(lV[buf ^ 1], pV);  async_copy16(lV[buf ^ 1] + 16, pV + 8);
    }

    // S = Q * K^T  (two 16x16 column tiles)
    v8f s0 = (v8f){0.f,0.f,0.f,0.f,0.f,0.f,0.f,0.f};
    v8f s1 = (v8f){0.f,0.f,0.f,0.f,0.f,0.f,0.f,0.f};
    const __bf16* Kb = &Ks[buf][0];
#pragma unroll
    for (int kc = 0; kc < 4; kc++) {
      s0 = wmma_bf16(qa[kc], fragB(Kb + col * 136 + kc * 32 + half * 16), s0);
      s1 = wmma_bf16(qa[kc], fragB(Kb + (16 + col) * 136 + kc * 32 + half * 16), s1);
    }
    // online softmax (per row r; rows r / r+8 live in lane halves)
#pragma unroll
    for (int r = 0; r < 8; r++) {
      float t = fmaxf(s0[r], s1[r]);
      for (int off = 8; off; off >>= 1) t = fmaxf(t, __shfl_xor(t, off));
      float mn = fmaxf(mi[r], t);
      float alpha = __expf(mi[r] - mn);
      mi[r] = mn;
      float p0 = __expf(s0[r] - mn), p1 = __expf(s1[r] - mn);
      s0[r] = p0; s1[r] = p1;
      float sum = p0 + p1;
      for (int off = 8; off; off >>= 1) sum += __shfl_xor(sum, off);
      li[r] = li[r] * alpha + sum;
#pragma unroll
      for (int t8 = 0; t8 < 8; t8++) o[t8][r] *= alpha;
    }
    // stage P (C-layout -> A-layout) through LDS
    __bf16* pw = Ps + w * 640;
#pragma unroll
    for (int r = 0; r < 8; r++) {
      pw[(r + 8 * half) * 40 + col] = (__bf16)s0[r];
      pw[(r + 8 * half) * 40 + 16 + col] = (__bf16)s1[r];
    }
    __syncthreads();
    v16bf pa = fragA(pw + col * 40, half);
    const __bf16* Vb = &Vs[buf][0];
#pragma unroll
    for (int t8 = 0; t8 < 8; t8++)
      o[t8] = wmma_bf16(pa, fragB(Vb + (t8 * 16 + col) * 40 + half * 16), o[t8]);
  }
  // normalize + write to cat[:, h*128 + d]
  float rinv[8];
#pragma unroll
  for (int r = 0; r < 8; r++) rinv[r] = 1.0f / li[r];
#pragma unroll
  for (int t8 = 0; t8 < 8; t8++) {
#pragma unroll
    for (int r = 0; r < 8; r++) {
      int lq = q0 + r + 8 * half;
      cat[(size_t)lq * K2 + h * DHEAD + t8 * 16 + col] =
          (__bf16)(o[t8][r] * rinv[r]);
    }
  }
}

// ---------------------------------------------------------------------------
// Kernel 6: GEMM2: out = x + gate * (cat @ lin2_w.T + b)
// ---------------------------------------------------------------------------
__global__ __launch_bounds__(256) void gemm2_kernel(
    const __bf16* __restrict__ A, const float* __restrict__ W,
    const float* __restrict__ bias, const float* __restrict__ x,
    const float* __restrict__ gate, float* __restrict__ out) {
  GEMM_MAINLOOP(K2)
#pragma unroll
  for (int i = 0; i < 4; i++) {
#pragma unroll
    for (int j = 0; j < 2; j++) {
      int n = nBase + wn * 32 + j * 16 + col;
      float bi = bias[n], gg = gate[n];
#pragma unroll
      for (int r = 0; r < 8; r++) {
        int l = mBase + wm * 64 + i * 16 + r + half * 8;
        float val = acc[i][j][r] + bi;
        out[(size_t)l * HID + n] = x[(size_t)l * HID + n] + gg * val;
      }
    }
  }
}

// ---------------------------------------------------------------------------
// Launcher
// ---------------------------------------------------------------------------
extern "C" void kernel_launch(void* const* d_in, const int* in_sizes, int n_in,
                              void* d_out, int out_size, void* d_ws,
                              size_t ws_size, hipStream_t stream) {
  const float* x       = (const float*)d_in[0];
  const float* vec     = (const float*)d_in[1];
  const float* pe      = (const float*)d_in[2];
  const float* mod_w   = (const float*)d_in[3];
  const float* mod_b   = (const float*)d_in[4];
  const float* lin1_w  = (const float*)d_in[5];
  const float* lin1_b  = (const float*)d_in[6];
  const float* lin2_w  = (const float*)d_in[7];
  const float* lin2_b  = (const float*)d_in[8];
  const float* q_scale = (const float*)d_in[9];
  const float* k_scale = (const float*)d_in[10];
  float* out = (float*)d_out;

  char* ws = (char*)d_ws;
  const size_t SZ_MOD  = (size_t)3 * HID * 4;            // 36 KB
  const size_t SZ_XMOD = (size_t)L_SEQ * HID * 2;        // 18.9 MB
  const size_t SZ_HEAD = (size_t)NHEAD * L_SEQ * DHEAD * 2;
  float*  mod  = (float*)ws;
  __bf16* xmod = (__bf16*)(ws + SZ_MOD);
  __bf16* qbuf = (__bf16*)(ws + SZ_MOD + SZ_XMOD);
  __bf16* kbuf = (__bf16*)(ws + SZ_MOD + SZ_XMOD + SZ_HEAD);
  __bf16* vbuf = (__bf16*)(ws + SZ_MOD + SZ_XMOD + 2 * SZ_HEAD);
  __bf16* cat  = (__bf16*)(ws + SZ_MOD + SZ_XMOD + 3 * SZ_HEAD);

  mod_gemv_kernel<<<dim3((3 * HID) / 8), 256, 0, stream>>>(vec, mod_w, mod_b, mod);
  ln_mod_kernel<<<dim3(L_SEQ), 256, 0, stream>>>(x, mod, xmod);
  // grid.x = M tiles so consecutive blocks share one 128-col weight panel (L2)
  gemm1_kernel<<<dim3(L_SEQ / 128, N1 / 128), 256, 0, stream>>>(
      xmod, lin1_w, lin1_b, qbuf, kbuf, vbuf, cat);
  rope_rms_kernel<<<dim3(L_SEQ, NHEAD, 2), 128, 0, stream>>>(
      qbuf, kbuf, pe, q_scale, k_scale);
  attn_kernel<<<dim3(L_SEQ / 128, NHEAD), 256, 0, stream>>>(qbuf, kbuf, vbuf, cat);
  gemm2_kernel<<<dim3(L_SEQ / 128, HID / 128), 256, 0, stream>>>(
      cat, lin2_w, lin2_b, x, mod + 2 * HID, out);
}
